// VariationalQuantumClassifier_55671366091164
// MI455X (gfx1250) — matrix operand order repaired and checked by reading, not compiled, not run on hardware
//
#include <hip/hip_runtime.h>
#include <math.h>

// ---------------------------------------------------------------------------
// Fused VQC forward for MI455X (gfx1250, wave32, WMMA), software-pipelined.
//   h   = relu(x @ W1 + b1)   16384x2048, K=1024  (split-bf16, 3-pass WMMA)
//   enc = tanh(h @ W2 + b2)   16384x16,   K=2048  (fused, h stays in LDS)
//   6 VQC layers + sigmoid head, per row
// Block = 64 batch rows, 256 threads (8 wave32s), grid = 256 blocks.
//
// If workspace permits (72.1 MB), a memory-bound pre-pass splits x/W1/W2 into
// bf16(hi,lo) (weights also pre-transposed). The hot loop then stages tiles
// with pure b128 load/store (no conversion VALU at all). Otherwise falls back
// to in-kernel splitting (same math).
// ---------------------------------------------------------------------------

typedef __attribute__((ext_vector_type(4)))  __bf16 v4bf;
typedef __attribute__((ext_vector_type(8)))  __bf16 v8bf;
typedef __attribute__((ext_vector_type(16))) __bf16 v16bf;
typedef __attribute__((ext_vector_type(8)))  float  v8f;
typedef __attribute__((ext_vector_type(4)))  float  fx4;

#define BATCH    16384
#define IN_DIM   1024
#define HIDDEN   2048
#define NQ       16
#define NLAYERS  6

#define BM 64            // batch rows per block
#define BN 64            // hidden cols per chunk
#define BKS 32           // K per pipelined stage (one WMMA k-step)
#define NKC (IN_DIM / BKS)
#define LDA 40           // bf16/row for A/B stage tiles (80 B: aligned + skewed)
#define LDH 72           // bf16/row for h / W2^T tiles (144 B)

// Shared arena (bytes, all 16B-aligned):
#define ASZ     (BM * LDA * 2)          // 5120 per array
#define BUFSZ   (4 * ASZ)               // 20480 per ping-pong buffer
#define OFF_HH  (2 * BUFSZ)             // 40960
#define OFF_HL  (OFF_HH + BM * LDH * 2) // 50176
#define OFF_W2H (OFF_HL + BM * LDH * 2) // 59392
#define OFF_W2L (OFF_W2H + NQ * LDH * 2)// 61696
#define SMEM_SZ (OFF_W2L + NQ * LDH * 2)// 64000 < 65536
// Tail overlays (stage buffers dead by then):
#define OFF_RED 0
#define OFF_CT  8192
#define OFF_SP  (OFF_CT + NLAYERS * NQ * 4)

// Workspace layout (bytes):
#define WS_XH   ((size_t)0)
#define WS_XL   (WS_XH + (size_t)BATCH * IN_DIM * 2)        // 33554432
#define WS_W1H  (WS_XL + (size_t)BATCH * IN_DIM * 2)        // 67108864
#define WS_W1L  (WS_W1H + (size_t)IN_DIM * HIDDEN * 2)      // 71303168
#define WS_W2H  (WS_W1L + (size_t)IN_DIM * HIDDEN * 2)      // 75497472
#define WS_W2L  (WS_W2H + (size_t)HIDDEN * NQ * 2)
#define WS_NEED (WS_W2L + (size_t)HIDDEN * NQ * 2)          // 75628544

__device__ __forceinline__ __bf16 bf_from_u16(unsigned short u) {
  union { unsigned short u; __bf16 b; } c; c.u = u; return c.b;
}

// f ~= hi + lo: hi = truncate-to-bf16(f) (exact), lo = RNE-bf16(f - hi).
__device__ __forceinline__ void split2bf(float f, __bf16 &hi, __bf16 &lo) {
  unsigned u  = __float_as_uint(f);
  unsigned hb = u & 0xFFFF0000u;
  hi = bf_from_u16((unsigned short)(hb >> 16));
  float rem = f - __uint_as_float(hb);
  unsigned ur = __float_as_uint(rem);
  unsigned r  = (ur + 0x7FFFu + ((ur >> 16) & 1u)) >> 16;
  lo = bf_from_u16((unsigned short)r);
}

// A fragment (16x32 MxK), bf16 LDS [row][k]:
// lanes 0-15: row=lane, K {0..7,16..23}+k0; lanes 16-31: same rows, K+8.
__device__ __forceinline__ v16bf frag_a(const __bf16* base, int rowbase, int k0,
                                        int stride, int lane) {
  const int m  = rowbase + (lane & 15);
  const int kb = k0 + ((lane & 16) ? 8 : 0);
  const __bf16* p = base + m * stride + kb;
  v8bf lo = *(const v8bf*)p;
  v8bf hi = *(const v8bf*)(p + 16);
  return __builtin_shufflevector(lo, hi, 0,1,2,3,4,5,6,7,8,9,10,11,12,13,14,15);
}

// B fragment (32x16 KxN), bf16 LDS transposed [n][k]:
// lanes 0-15: col=lane, K {0..7,16..23}+k0; lanes 16-31: same cols, K+8.
__device__ __forceinline__ v16bf frag_b(const __bf16* base, int colbase, int k0,
                                        int stride, int lane) {
  const int n  = colbase + (lane & 15);
  const int kb = k0 + ((lane & 16) ? 8 : 0);
  const __bf16* p = base + n * stride + kb;
  v8bf lo = *(const v8bf*)p;
  v8bf hi = *(const v8bf*)(p + 16);
  return __builtin_shufflevector(lo, hi, 0,1,2,3,4,5,6,7,8,9,10,11,12,13,14,15);
}

// 3-pass split-bf16 product: acc += Ahi*Bhi + Ahi*Blo + Alo*Bhi (f32 accum).
__device__ __forceinline__ v8f wmma3(v16bf ahi, v16bf alo, v16bf bhi, v16bf blo, v8f acc) {
  acc = __builtin_amdgcn_wmma_f32_16x16x32_bf16(false, alo, false, bhi, (short)0, acc, false, false);
  acc = __builtin_amdgcn_wmma_f32_16x16x32_bf16(false, ahi, false, blo, (short)0, acc, false, false);
  acc = __builtin_amdgcn_wmma_f32_16x16x32_bf16(false, ahi, false, bhi, (short)0, acc, false, false);
  return acc;
}

struct StageRegs {
  fx4 a0, a1; float b[8];      // f32 fallback path
  v8bf ah, al, bh, bl;         // pre-split path
};

template <bool PS>
__device__ __forceinline__ void stage_load(StageRegs &r,
    const float* __restrict__ x, const float* __restrict__ W1,
    const __bf16* __restrict__ xh, const __bf16* __restrict__ xl,
    const __bf16* __restrict__ w1th, const __bf16* __restrict__ w1tl,
    int m0, int n0, int kg, int tid) {
  if (PS) {
    const int mm = tid >> 2, kk = (tid & 3) * 8;     // 64 rows x 4 v8bf
    r.ah = *(const v8bf*)(xh   + (size_t)(m0 + mm) * IN_DIM + kg + kk);
    r.al = *(const v8bf*)(xl   + (size_t)(m0 + mm) * IN_DIM + kg + kk);
    r.bh = *(const v8bf*)(w1th + (size_t)(n0 + mm) * IN_DIM + kg + kk);
    r.bl = *(const v8bf*)(w1tl + (size_t)(n0 + mm) * IN_DIM + kg + kk);
  } else {
    const int i0 = tid, i1 = tid + 256;
    r.a0 = *(const fx4*)(x + (size_t)(m0 + (i0 >> 3)) * IN_DIM + kg + (i0 & 7) * 4);
    r.a1 = *(const fx4*)(x + (size_t)(m0 + (i1 >> 3)) * IN_DIM + kg + (i1 & 7) * 4);
#pragma unroll
    for (int t = 0; t < 8; ++t) {
      const int i = tid + t * 256;
      r.b[t] = W1[(size_t)(kg + (i >> 6)) * HIDDEN + n0 + (i & 63)];
    }
  }
}

template <bool PS>
__device__ __forceinline__ void stage_store(const StageRegs &r, unsigned char* smem,
                                            int buf, int tid) {
  __bf16* Ah = (__bf16*)(smem + buf * BUFSZ);
  __bf16* Al = (__bf16*)(smem + buf * BUFSZ + ASZ);
  __bf16* Bh = (__bf16*)(smem + buf * BUFSZ + 2 * ASZ);
  __bf16* Bl = (__bf16*)(smem + buf * BUFSZ + 3 * ASZ);
  if (PS) {
    const int mm = tid >> 2, kk = (tid & 3) * 8;
    *(v8bf*)(Ah + mm * LDA + kk) = r.ah;
    *(v8bf*)(Al + mm * LDA + kk) = r.al;
    *(v8bf*)(Bh + mm * LDA + kk) = r.bh;
    *(v8bf*)(Bl + mm * LDA + kk) = r.bl;
  } else {
    const int i0 = tid, i1 = tid + 256;
    v4bf h4, l4;
#pragma unroll
    for (int e = 0; e < 4; ++e) { __bf16 h, l; split2bf(r.a0[e], h, l); h4[e] = h; l4[e] = l; }
    *(v4bf*)(Ah + (i0 >> 3) * LDA + (i0 & 7) * 4) = h4;
    *(v4bf*)(Al + (i0 >> 3) * LDA + (i0 & 7) * 4) = l4;
#pragma unroll
    for (int e = 0; e < 4; ++e) { __bf16 h, l; split2bf(r.a1[e], h, l); h4[e] = h; l4[e] = l; }
    *(v4bf*)(Ah + (i1 >> 3) * LDA + (i1 & 7) * 4) = h4;
    *(v4bf*)(Al + (i1 >> 3) * LDA + (i1 & 7) * 4) = l4;
#pragma unroll
    for (int t = 0; t < 8; ++t) {
      const int i = tid + t * 256;
      const int kk = i >> 6, nn = i & 63;
      __bf16 h, l; split2bf(r.b[t], h, l);
      Bh[nn * LDA + kk] = h;
      Bl[nn * LDA + kk] = l;
    }
  }
}

template <bool PS>
__global__ __launch_bounds__(256)
void vqc_fused_kernel(const float* __restrict__ x,  const float* __restrict__ W1,
                      const float* __restrict__ b1, const float* __restrict__ W2,
                      const float* __restrict__ b2, const float* __restrict__ vqc,
                      const float* __restrict__ Wd, const float* __restrict__ bd,
                      const __bf16* __restrict__ xh,   const __bf16* __restrict__ xl,
                      const __bf16* __restrict__ w1th, const __bf16* __restrict__ w1tl,
                      const __bf16* __restrict__ w2th, const __bf16* __restrict__ w2tl,
                      float* __restrict__ out) {
  __shared__ __align__(16) unsigned char smem[SMEM_SZ];
  __bf16* Hh  = (__bf16*)(smem + OFF_HH);
  __bf16* Hl  = (__bf16*)(smem + OFF_HL);
  __bf16* W2h = (__bf16*)(smem + OFF_W2H);
  __bf16* W2l = (__bf16*)(smem + OFF_W2L);

  const int tid  = threadIdx.x;
  const int lane = tid & 31;
  const int w    = tid >> 5;
  const int wr   = w & 3;        // 16-row tile
  const int wh   = w >> 2;       // 0/1: GEMM1 col-half, GEMM2 k-half
  const int m0   = blockIdx.x * BM;

  v8f enc = {};

  for (int nc = 0; nc < HIDDEN / BN; ++nc) {
    const int n0 = nc * BN;
    v8f acc0 = {}, acc1 = {};

    StageRegs sr;
    stage_load<PS>(sr, x, W1, xh, xl, w1th, w1tl, m0, n0, 0, tid);
    stage_store<PS>(sr, smem, 0, tid);
    __syncthreads();

#pragma unroll 2
    for (int kc = 0; kc < NKC; ++kc) {
      const int cur = kc & 1;
      const bool hasNext = (kc + 1 < NKC);
      if (hasNext) stage_load<PS>(sr, x, W1, xh, xl, w1th, w1tl, m0, n0, (kc + 1) * BKS, tid);

      const __bf16* Ah = (const __bf16*)(smem + cur * BUFSZ);
      const __bf16* Al = (const __bf16*)(smem + cur * BUFSZ + ASZ);
      const __bf16* Bh = (const __bf16*)(smem + cur * BUFSZ + 2 * ASZ);
      const __bf16* Bl = (const __bf16*)(smem + cur * BUFSZ + 3 * ASZ);
      v16bf ahi = frag_a(Ah, wr * 16, 0, LDA, lane);
      v16bf alo = frag_a(Al, wr * 16, 0, LDA, lane);
      v16bf bhi = frag_b(Bh, (wh * 2 + 0) * 16, 0, LDA, lane);
      v16bf blo = frag_b(Bl, (wh * 2 + 0) * 16, 0, LDA, lane);
      acc0 = wmma3(ahi, alo, bhi, blo, acc0);
      bhi = frag_b(Bh, (wh * 2 + 1) * 16, 0, LDA, lane);
      blo = frag_b(Bl, (wh * 2 + 1) * 16, 0, LDA, lane);
      acc1 = wmma3(ahi, alo, bhi, blo, acc1);

      if (hasNext) stage_store<PS>(sr, smem, cur ^ 1, tid);
      __syncthreads();
    }

    // bias + relu, split once (data-dependent), spill h chunk as bf16 hi/lo
    {
      const int half8 = (lane & 16) ? 8 : 0;
      const int c0 = (wh * 2 + 0) * 16 + (lane & 15);
      const int c1 = (wh * 2 + 1) * 16 + (lane & 15);
      const float bv0 = b1[n0 + c0];
      const float bv1 = b1[n0 + c1];
#pragma unroll
      for (int r = 0; r < 8; ++r) {
        const int m = wr * 16 + half8 + r;
        __bf16 h, l;
        split2bf(fmaxf(acc0[r] + bv0, 0.f), h, l);
        Hh[m * LDH + c0] = h; Hl[m * LDH + c0] = l;
        split2bf(fmaxf(acc1[r] + bv1, 0.f), h, l);
        Hh[m * LDH + c1] = h; Hl[m * LDH + c1] = l;
      }
    }
    // stage W2 chunk transposed [q][n_local]
    if (PS) {
      if (tid < 128) {
        const int q = tid >> 3, kk = (tid & 7) * 8;
        *(v8bf*)(W2h + q * LDH + kk) = *(const v8bf*)(w2th + (size_t)q * HIDDEN + n0 + kk);
      } else {
        const int t = tid - 128;
        const int q = t >> 3, kk = (t & 7) * 8;
        *(v8bf*)(W2l + q * LDH + kk) = *(const v8bf*)(w2tl + (size_t)q * HIDDEN + n0 + kk);
      }
    } else {
      for (int i = tid; i < BN * NQ; i += 256) {
        const int k = i >> 4, q = i & 15;
        __bf16 h, l; split2bf(W2[(size_t)(n0 + k) * NQ + q], h, l);
        W2h[q * LDH + k] = h;
        W2l[q * LDH + k] = l;
      }
    }
    __syncthreads();

    // fused GEMM2 partial: enc += h[:, wh*32 : wh*32+32] @ W2_chunk
    {
      v16bf ahi = frag_a(Hh, wr * 16, wh * 32, LDH, lane);
      v16bf alo = frag_a(Hl, wr * 16, wh * 32, LDH, lane);
      v16bf bhi = frag_b(W2h, 0, wh * 32, LDH, lane);
      v16bf blo = frag_b(W2l, 0, wh * 32, LDH, lane);
      enc = wmma3(ahi, alo, bhi, blo, enc);
    }
    __syncthreads();
  }

  // tail phase: overlay dead stage buffers
  float* red = (float*)(smem + OFF_RED);
  float* ct  = (float*)(smem + OFF_CT);
  float* sp  = (float*)(smem + OFF_SP);
  {
    const int half8 = (lane & 16) ? 8 : 0;
#pragma unroll
    for (int r = 0; r < 8; ++r)
      red[((wh * BM) + wr * 16 + half8 + r) * NQ + (lane & 15)] = enc[r];
  }
  if (tid < NLAYERS * NQ) {
    const float th = vqc[tid * 2 + 0];
    const float ph = vqc[tid * 2 + 1];
    ct[tid] = cosf(th);
    sp[tid] = sinf(th) * cosf(ph);
  }
  __syncthreads();

  if (tid < BM) {
    float s[NQ];
#pragma unroll
    for (int q = 0; q < NQ; ++q) {
      const float v = red[tid * NQ + q] + red[(BM + tid) * NQ + q] + b2[q];
      s[q] = sinf(tanhf(v) * 3.14159265358979323846f);
    }
#pragma unroll
    for (int layer = 0; layer < NLAYERS; ++layer) {
#pragma unroll
      for (int q = 0; q < NQ; ++q)
        s[q] = ct[layer * NQ + q] * s[q] + sp[layer * NQ + q];
      if (layer < NLAYERS - 1) {
#pragma unroll
        for (int e2 = 0; e2 < NQ / 2; ++e2)
          s[2 * e2 + 1] = s[2 * e2] * 0.1f + s[2 * e2 + 1] * 0.9f;
      }
    }
    float acc = bd[0];
#pragma unroll
    for (int q = 0; q < NQ; ++q) acc += s[q] * Wd[q];
    out[m0 + tid] = 1.f / (1.f + expf(-acc));
  }
}

// --- pre-pass kernels (memory-bound, run once per launch) ---

__global__ __launch_bounds__(256)
void presplit_x_kernel(const float* __restrict__ src, __bf16* __restrict__ hi,
                       __bf16* __restrict__ lo, int n4) {
  const int i = blockIdx.x * 256 + threadIdx.x;
  if (i >= n4) return;
  fx4 v = *(const fx4*)(src + (size_t)i * 4);
  v4bf h4, l4;
#pragma unroll
  for (int e = 0; e < 4; ++e) { __bf16 h, l; split2bf(v[e], h, l); h4[e] = h; l4[e] = l; }
  *(v4bf*)(hi + (size_t)i * 4) = h4;
  *(v4bf*)(lo + (size_t)i * 4) = l4;
}

// src[k][n] (rows x cols, row-major) -> hi/lo[n][k]
__global__ __launch_bounds__(256)
void presplit_transpose_kernel(const float* __restrict__ src, __bf16* __restrict__ hi,
                               __bf16* __restrict__ lo, int rows, int cols) {
  const int i = blockIdx.x * 256 + threadIdx.x;
  if (i >= rows * cols) return;
  const int k = i / cols, n = i % cols;   // coalesced read
  __bf16 h, l; split2bf(src[i], h, l);
  hi[(size_t)n * rows + k] = h;
  lo[(size_t)n * rows + k] = l;
}

extern "C" void kernel_launch(void* const* d_in, const int* in_sizes, int n_in,
                              void* d_out, int out_size, void* d_ws, size_t ws_size,
                              hipStream_t stream) {
  (void)in_sizes; (void)n_in; (void)out_size;
  const float* x  = (const float*)d_in[0];
  const float* W1 = (const float*)d_in[1];
  const float* b1 = (const float*)d_in[2];
  const float* W2 = (const float*)d_in[3];
  const float* b2 = (const float*)d_in[4];
  const float* vq = (const float*)d_in[5];
  const float* Wd = (const float*)d_in[6];
  const float* bd = (const float*)d_in[7];
  float* out = (float*)d_out;

  dim3 grid(BATCH / BM);   // 256 blocks
  dim3 block(256);         // 8 wave32s

  if (d_ws != nullptr && ws_size >= WS_NEED) {
    unsigned char* ws = (unsigned char*)d_ws;
    __bf16* xh   = (__bf16*)(ws + WS_XH);
    __bf16* xl   = (__bf16*)(ws + WS_XL);
    __bf16* w1th = (__bf16*)(ws + WS_W1H);
    __bf16* w1tl = (__bf16*)(ws + WS_W1L);
    __bf16* w2th = (__bf16*)(ws + WS_W2H);
    __bf16* w2tl = (__bf16*)(ws + WS_W2L);

    const int xn4 = BATCH * IN_DIM / 4;
    hipLaunchKernelGGL(presplit_x_kernel, dim3((xn4 + 255) / 256), dim3(256), 0, stream,
                       x, xh, xl, xn4);
    hipLaunchKernelGGL(presplit_transpose_kernel,
                       dim3((IN_DIM * HIDDEN + 255) / 256), dim3(256), 0, stream,
                       W1, w1th, w1tl, IN_DIM, HIDDEN);
    hipLaunchKernelGGL(presplit_transpose_kernel,
                       dim3((HIDDEN * NQ + 255) / 256), dim3(256), 0, stream,
                       W2, w2th, w2tl, HIDDEN, NQ);
    hipLaunchKernelGGL((vqc_fused_kernel<true>), grid, block, 0, stream,
                       x, W1, b1, W2, b2, vq, Wd, bd,
                       xh, xl, w1th, w1tl, w2th, w2tl, out);
  } else {
    hipLaunchKernelGGL((vqc_fused_kernel<false>), grid, block, 0, stream,
                       x, W1, b1, W2, b2, vq, Wd, bd,
                       (const __bf16*)nullptr, (const __bf16*)nullptr,
                       (const __bf16*)nullptr, (const __bf16*)nullptr,
                       (const __bf16*)nullptr, (const __bf16*)nullptr, out);
  }
}
